// TempME_TGAT_22308060135613
// MI455X (gfx1250) — matrix operand order repaired and checked by reading, not compiled, not run on hardware
//
#include <hip/hip_runtime.h>
#include <hip/hip_bf16.h>
#include <math.h>

// ---------------------------------------------------------------------------
// Types for CDNA5 WMMA (gfx1250, wave32)
// ---------------------------------------------------------------------------
typedef __bf16 bf16;
typedef __attribute__((ext_vector_type(16))) __bf16 v16bf;
typedef __attribute__((ext_vector_type(8)))  __bf16 v8bf;
typedef __attribute__((ext_vector_type(8)))  float  v8f;

#define NT 19200   // B*W*L tokens for cat encoder
#define NS 6400    // B*W sequences
#define D0 688     // GRU dim
#define DP 704     // padded to /32
#define QKV0 2064
#define QKVP 2112
#define FFND 4096
#define FP0 472    // final feature dim
#define FPP 480

__device__ __forceinline__ bf16 to_bf16(float x) { return (bf16)x; }

// ---------------------------------------------------------------------------
// CDNA5 async global->LDS copy (bypasses VGPRs, tracked by ASYNCcnt)
// ---------------------------------------------------------------------------
__device__ __forceinline__ unsigned lds_addr32(const void* p) {
    // generic -> LDS address space (3); AS(3) pointers are 32-bit LDS offsets
    return (unsigned)(size_t)(const __attribute__((address_space(3))) char*)p;
}

__device__ __forceinline__ void async_copy_b128(void* lds_dst, const void* gsrc) {
    asm volatile("global_load_async_to_lds_b128 %0, %1, off"
                 :: "v"(lds_addr32(lds_dst)), "v"(gsrc)
                 : "memory");
}

__device__ __forceinline__ void wait_async0() {
    asm volatile("s_wait_asynccnt 0x0" ::: "memory");
}

// ---------------------------------------------------------------------------
// Generic GEMM: out[M,N] = A[M,K] @ W[N,K]^T + bias, optional relu.
// A, W bf16 (K padded to /32, N padded to /64, M multiple of 128).
// Writes fp32 and/or bf16 outputs (row-major, ld = N).
// Block = 256 threads = 8 waves; tile 128x64; each wave 32x32 via 2x2 WMMA.
// LDS tiles double-buffered and filled with global_load_async_to_lds_b128.
// ---------------------------------------------------------------------------
__global__ __launch_bounds__(256)
void gemm_bf16(const bf16* __restrict__ A, const bf16* __restrict__ Wt,
               const float* __restrict__ bias,
               float* __restrict__ outF, bf16* __restrict__ outB,
               int M, int N, int K, int relu)
{
    __shared__ __align__(16) bf16 As[2][128][32];
    __shared__ __align__(16) bf16 Bs[2][64][32];

    const int tid  = threadIdx.x;
    const int wave = tid >> 5;
    const int lane = tid & 31;
    const int wm   = wave >> 1;     // 0..3 -> 32-row strip
    const int wn   = wave & 1;      // 0..1 -> 32-col strip
    const int bm   = blockIdx.x * 128;
    const int bn   = blockIdx.y * 64;

    const int lrow = tid >> 2;          // 0..63
    const int lk   = (tid & 3) * 8;     // 0,8,16,24

    const int m16 = lane & 15;
    const int hi  = lane >> 4;

    // per-thread global source pointers (advance by 64B per k-step)
    const bf16* ga0 = &A[(size_t)(bm + lrow) * K + lk];
    const bf16* ga1 = &A[(size_t)(bm + lrow + 64) * K + lk];
    const bf16* gb0 = &Wt[(size_t)(bn + lrow) * K + lk];

    v8f acc[2][2];
    #pragma unroll
    for (int i = 0; i < 2; ++i)
        #pragma unroll
        for (int j = 0; j < 2; ++j)
            acc[i][j] = (v8f){0.f,0.f,0.f,0.f,0.f,0.f,0.f,0.f};

    // preload tile 0 into buffer 0
    async_copy_b128(&As[0][lrow][lk],      ga0);
    async_copy_b128(&As[0][lrow + 64][lk], ga1);
    async_copy_b128(&Bs[0][lrow][lk],      gb0);
    wait_async0();
    __syncthreads();

    int cur = 0;
    for (int k0 = 0; k0 < K; k0 += 32) {
        // prefetch next k-tile into the other buffer while computing
        if (k0 + 32 < K) {
            const int nxt = cur ^ 1;
            async_copy_b128(&As[nxt][lrow][lk],      ga0 + k0 + 32);
            async_copy_b128(&As[nxt][lrow + 64][lk], ga1 + k0 + 32);
            async_copy_b128(&Bs[nxt][lrow][lk],      gb0 + k0 + 32);
        }

        #pragma unroll
        for (int mi = 0; mi < 2; ++mi) {
            // A fragment: 16x32 bf16, lane m = lane%16, K split per ISA layout
            const int ar = wm * 32 + mi * 16 + m16;
            v8bf alo = *(const v8bf*)&As[cur][ar][8 * hi];       // K = 8*hi..
            v8bf ahi = *(const v8bf*)&As[cur][ar][16 + 8 * hi];  // K = 16+8*hi..
            v16bf a = __builtin_shufflevector(alo, ahi,
                0,1,2,3,4,5,6,7,8,9,10,11,12,13,14,15);
            #pragma unroll
            for (int ni = 0; ni < 2; ++ni) {
                // B fragment: 32x16; lane n = lane%16, K split per ISA layout (same as A)
                const int br = wn * 32 + ni * 16 + m16;
                v8bf blo = *(const v8bf*)&Bs[cur][br][8 * hi];
                v8bf bhi = *(const v8bf*)&Bs[cur][br][16 + 8 * hi];
                v16bf b = __builtin_shufflevector(blo, bhi,
                    0,1,2,3,4,5,6,7,8,9,10,11,12,13,14,15);
                acc[mi][ni] = __builtin_amdgcn_wmma_f32_16x16x32_bf16(
                    false, a, false, b, (short)0, acc[mi][ni], false, false);
            }
        }

        wait_async0();      // my async loads into nxt buffer have landed
        __syncthreads();    // everyone done reading cur + loading nxt
        cur ^= 1;
    }

    // Epilogue: C layout -> lane n = lane%16 (col), VGPR r -> row r + 8*hi
    #pragma unroll
    for (int mi = 0; mi < 2; ++mi) {
        #pragma unroll
        for (int ni = 0; ni < 2; ++ni) {
            const int col = bn + wn * 32 + ni * 16 + m16;
            const float bv = bias ? bias[col] : 0.f;
            #pragma unroll
            for (int r = 0; r < 8; ++r) {
                const int row = bm + wm * 32 + mi * 16 + r + 8 * hi;
                float v = acc[mi][ni][r] + bv;
                if (relu) v = fmaxf(v, 0.f);
                if (outF) outF[(size_t)row * N + col] = v;
                if (outB) outB[(size_t)row * N + col] = to_bf16(v);
            }
        }
    }
}

// ---------------------------------------------------------------------------
// Weight prepack: W[N,K] f32 -> Wp[Np,Kp] bf16 zero padded
// ---------------------------------------------------------------------------
__global__ __launch_bounds__(256)
void pack_w(const float* __restrict__ Wsrc, bf16* __restrict__ Wp,
            int N, int K, int Kp)
{
    const int n = blockIdx.x;   // 0..Np-1
    for (int k = threadIdx.x; k < Kp; k += 256) {
        float v = 0.f;
        if (n < N && k < K) v = Wsrc[(size_t)n * K + k];
        Wp[(size_t)n * Kp + k] = to_bf16(v);
    }
}

__global__ __launch_bounds__(256)
void pack_bias(const float* __restrict__ b, float* __restrict__ bp, int N, int Np)
{
    const int i = blockIdx.x * 256 + threadIdx.x;
    if (i < Np) bp[i] = (i < N) ? b[i] : 0.f;
}

// ---------------------------------------------------------------------------
// Build X[NT, 688] fp32 + Xb[NT, 704] bf16 (gather + time cosine encoding)
// token t = ((b*100 + w)*3 + l)
// ---------------------------------------------------------------------------
__global__ __launch_bounds__(256)
void build_x(const float* __restrict__ node_table, const float* __restrict__ edge_table,
             const float* __restrict__ basis_freq, const float* __restrict__ phase,
             const int* __restrict__ node_idx, const int* __restrict__ edge_idx,
             const float* __restrict__ time_idx,
             float* __restrict__ X32, bf16* __restrict__ Xb)
{
    const int t = blockIdx.x;
    const int l = t % 3;
    const int s = t / 3;
    const int e    = edge_idx[s * 3 + l];
    const int nsrc = node_idx[s * 6 + 2 * l];
    const int ntgt = node_idx[s * 6 + 2 * l + 1];
    const float dt = time_idx[s * 3 + 2] - time_idx[s * 3 + l];

    for (int c = threadIdx.x; c < DP; c += 256) {
        float v = 0.f;
        if (c < 172)      v = edge_table[(size_t)e * 172 + c];
        else if (c < 344) { int j = c - 172; v = cosf(dt * basis_freq[j] + phase[j]); }
        else if (c < 516) v = node_table[(size_t)nsrc * 172 + (c - 344)];
        else if (c < D0)  v = node_table[(size_t)ntgt * 172 + (c - 516)];
        if (c < D0) X32[(size_t)t * D0 + c] = v;
        Xb[(size_t)t * DP + c] = to_bf16(v);
    }
}

// ---------------------------------------------------------------------------
// cat attention: L=3, hd=86, one wave per (seq, head). qkv fp32 ld=2112.
// Writes bf16 ao[NT, 704] (head h==0 also zeroes pad cols 688..703)
// ---------------------------------------------------------------------------
__global__ __launch_bounds__(32)
void cat_attn(const float* __restrict__ qkv, bf16* __restrict__ ao)
{
    const int g = blockIdx.x;           // 0..51199
    const int s = g >> 3, h = g & 7;
    const int lane = threadIdx.x;
    const float* base = qkv + (size_t)s * 3 * QKVP;
    const int co = h * 86;

    float dots[9];
    #pragma unroll
    for (int i = 0; i < 9; ++i) dots[i] = 0.f;

    for (int d = lane; d < 86; d += 32) {
        float q0 = base[0 * QKVP + co + d];
        float q1 = base[1 * QKVP + co + d];
        float q2 = base[2 * QKVP + co + d];
        float k0 = base[0 * QKVP + D0 + co + d];
        float k1 = base[1 * QKVP + D0 + co + d];
        float k2 = base[2 * QKVP + D0 + co + d];
        dots[0] += q0 * k0; dots[1] += q0 * k1; dots[2] += q0 * k2;
        dots[3] += q1 * k0; dots[4] += q1 * k1; dots[5] += q1 * k2;
        dots[6] += q2 * k0; dots[7] += q2 * k1; dots[8] += q2 * k2;
    }
    #pragma unroll
    for (int i = 0; i < 9; ++i)
        #pragma unroll
        for (int off = 16; off > 0; off >>= 1)
            dots[i] += __shfl_xor(dots[i], off, 32);

    const float scale = rsqrtf(86.f);
    float a[3][3];
    #pragma unroll
    for (int i = 0; i < 3; ++i) {
        float s0 = dots[i*3+0]*scale, s1 = dots[i*3+1]*scale, s2 = dots[i*3+2]*scale;
        float mx = fmaxf(s0, fmaxf(s1, s2));
        float e0 = expf(s0 - mx), e1 = expf(s1 - mx), e2 = expf(s2 - mx);
        float inv = 1.f / (e0 + e1 + e2);
        a[i][0] = e0 * inv; a[i][1] = e1 * inv; a[i][2] = e2 * inv;
    }
    for (int d = lane; d < 86; d += 32) {
        float v0 = base[0 * QKVP + 2 * D0 + co + d];
        float v1 = base[1 * QKVP + 2 * D0 + co + d];
        float v2 = base[2 * QKVP + 2 * D0 + co + d];
        #pragma unroll
        for (int i = 0; i < 3; ++i) {
            float o = a[i][0] * v0 + a[i][1] * v1 + a[i][2] * v2;
            ao[(size_t)(s * 3 + i) * DP + co + d] = to_bf16(o);
        }
    }
    if (h == 0) {   // zero pad columns once per seq
        for (int c = D0 + lane; c < DP; c += 32)
            #pragma unroll
            for (int i = 0; i < 3; ++i)
                ao[(size_t)(s * 3 + i) * DP + c] = to_bf16(0.f);
    }
}

// ---------------------------------------------------------------------------
// Add + LayerNorm: y = g*(a+b - mean)/sqrt(var+eps) + beta
// Writes fp32 (ld=D) and optional padded bf16 (ld=ldbb, pad cols zeroed)
// ---------------------------------------------------------------------------
__global__ __launch_bounds__(256)
void add_ln(const float* __restrict__ a, int lda,
            const float* __restrict__ b, int ldb,
            const float* __restrict__ gamma, const float* __restrict__ beta,
            int D, float* __restrict__ outF, bf16* __restrict__ outB, int ldbb)
{
    __shared__ float sh[256];
    const int row = blockIdx.x;
    const int tid = threadIdx.x;

    float xv[4];
    int cnt = 0;
    float s = 0.f;
    for (int c = tid; c < D; c += 256) {
        float v = a[(size_t)row * lda + c] + b[(size_t)row * ldb + c];
        xv[cnt++] = v;
        s += v;
    }
    sh[tid] = s; __syncthreads();
    for (int st = 128; st > 0; st >>= 1) { if (tid < st) sh[tid] += sh[tid + st]; __syncthreads(); }
    const float mean = sh[0] / (float)D;
    __syncthreads();

    float s2 = 0.f;
    for (int i = 0; i < cnt; ++i) { float d = xv[i] - mean; s2 += d * d; }
    sh[tid] = s2; __syncthreads();
    for (int st = 128; st > 0; st >>= 1) { if (tid < st) sh[tid] += sh[tid + st]; __syncthreads(); }
    const float inv = rsqrtf(sh[0] / (float)D + 1e-5f);

    cnt = 0;
    for (int c = tid; c < D; c += 256) {
        float y = gamma[c] * (xv[cnt++] - mean) * inv + beta[c];
        if (outF) outF[(size_t)row * D + c] = y;
        if (outB) outB[(size_t)row * ldbb + c] = to_bf16(y);
    }
    if (outB) {
        for (int c = D + tid; c < ldbb; c += 256)
            outB[(size_t)row * ldbb + c] = to_bf16(0.f);
    }
}

// ---------------------------------------------------------------------------
// Mean over L=3 -> padded bf16 [NS, 704]
// ---------------------------------------------------------------------------
__global__ __launch_bounds__(256)
void mean3(const float* __restrict__ h2, bf16* __restrict__ encb)
{
    const int n = blockIdx.x;
    for (int c = threadIdx.x; c < DP; c += 256) {
        float v = 0.f;
        if (c < D0) {
            v = (h2[(size_t)(3 * n + 0) * D0 + c] +
                 h2[(size_t)(3 * n + 1) * D0 + c] +
                 h2[(size_t)(3 * n + 2) * D0 + c]) * (1.f / 3.f);
        }
        encb[(size_t)n * DP + c] = to_bf16(v);
    }
}

// ---------------------------------------------------------------------------
// sa attention: seq=100, hd=16, block per (batch, head). sqkv fp32 ld=384.
// ---------------------------------------------------------------------------
__global__ __launch_bounds__(128)
void sa_attn(const float* __restrict__ sqkv, bf16* __restrict__ sao)
{
    __shared__ float Ks[100][16];
    __shared__ float Vs[100][16];
    const int g = blockIdx.x;           // 0..511
    const int b = g >> 3, h = g & 7;
    const int tid = threadIdx.x;

    for (int i = tid; i < 1600; i += 128) {
        const int w = i >> 4, d = i & 15;
        const size_t row = (size_t)(b * 100 + w) * 384;
        Ks[w][d] = sqkv[row + 128 + h * 16 + d];
        Vs[w][d] = sqkv[row + 256 + h * 16 + d];
    }
    __syncthreads();

    for (int qi = tid; qi < 100; qi += 128) {
        float q[16];
        const size_t qrow = (size_t)(b * 100 + qi) * 384 + h * 16;
        #pragma unroll
        for (int d = 0; d < 16; ++d) q[d] = sqkv[qrow + d];

        float mx = -3.4e38f;
        for (int j = 0; j < 100; ++j) {
            float sc = 0.f;
            #pragma unroll
            for (int d = 0; d < 16; ++d) sc += q[d] * Ks[j][d];
            mx = fmaxf(mx, sc * 0.25f);
        }
        float num[16];
        #pragma unroll
        for (int d = 0; d < 16; ++d) num[d] = 0.f;
        float den = 0.f;
        for (int j = 0; j < 100; ++j) {
            float sc = 0.f;
            #pragma unroll
            for (int d = 0; d < 16; ++d) sc += q[d] * Ks[j][d];
            const float e = expf(sc * 0.25f - mx);
            den += e;
            #pragma unroll
            for (int d = 0; d < 16; ++d) num[d] += e * Vs[j][d];
        }
        const float inv = 1.f / den;
        const size_t orow = (size_t)(b * 100 + qi) * 128 + h * 16;
        #pragma unroll
        for (int d = 0; d < 16; ++d) sao[orow + d] = to_bf16(num[d] * inv);
    }
}

// ---------------------------------------------------------------------------
// Build final features [NS, 480] bf16: [g(128) | src_node(172) | tgt_node(172) | pad]
// ---------------------------------------------------------------------------
__global__ __launch_bounds__(128)
void build_feats(const float* __restrict__ gfin, const float* __restrict__ node_table,
                 const int* __restrict__ src_idx, const int* __restrict__ tgt_idx,
                 bf16* __restrict__ feats)
{
    const int r = blockIdx.x;        // 0..NS-1
    const int b = r / 100;
    const int si = src_idx[b];
    const int ti = tgt_idx[b];
    for (int c = threadIdx.x; c < FPP; c += 128) {
        float v = 0.f;
        if (c < 128)      v = gfin[(size_t)r * 128 + c];
        else if (c < 300) v = node_table[(size_t)si * 172 + (c - 128)];
        else if (c < FP0) v = node_table[(size_t)ti * 172 + (c - 300)];
        feats[(size_t)r * FPP + c] = to_bf16(v);
    }
}

// ---------------------------------------------------------------------------
// Final head: sigmoid(ff[r,:] . W2 + b2) -> out[r]
// ---------------------------------------------------------------------------
__global__ __launch_bounds__(128)
void final_head(const float* __restrict__ ff, const float* __restrict__ W2,
                const float* __restrict__ b2, float* __restrict__ out)
{
    __shared__ float sh[128];
    const int r = blockIdx.x;
    const int tid = threadIdx.x;
    sh[tid] = ff[(size_t)r * 128 + tid] * W2[tid];
    __syncthreads();
    for (int st = 64; st > 0; st >>= 1) { if (tid < st) sh[tid] += sh[tid + st]; __syncthreads(); }
    if (tid == 0) out[r] = 1.f / (1.f + expf(-(sh[0] + b2[0])));
}

// ---------------------------------------------------------------------------
// Host orchestration
// ---------------------------------------------------------------------------
extern "C" void kernel_launch(void* const* d_in, const int* in_sizes, int n_in,
                              void* d_out, int out_size, void* d_ws, size_t ws_size,
                              hipStream_t stream)
{
    (void)in_sizes; (void)n_in; (void)out_size; (void)ws_size;

    const float* node_table = (const float*)d_in[0];
    const float* edge_table = (const float*)d_in[1];
    const float* basis_freq = (const float*)d_in[2];
    const float* phase      = (const float*)d_in[3];
    const float* cat_Wqkv = (const float*)d_in[4];
    const float* cat_bqkv = (const float*)d_in[5];
    const float* cat_Wo   = (const float*)d_in[6];
    const float* cat_bo   = (const float*)d_in[7];
    const float* cat_g1   = (const float*)d_in[8];
    const float* cat_b1   = (const float*)d_in[9];
    const float* cat_W1   = (const float*)d_in[10];
    const float* cat_c1   = (const float*)d_in[11];
    const float* cat_W2   = (const float*)d_in[12];
    const float* cat_c2   = (const float*)d_in[13];
    const float* cat_g2   = (const float*)d_in[14];
    const float* cat_b2   = (const float*)d_in[15];
    const float* sa_Wqkv  = (const float*)d_in[16];
    const float* sa_bqkv  = (const float*)d_in[17];
    const float* sa_Wo    = (const float*)d_in[18];
    const float* sa_bo    = (const float*)d_in[19];
    const float* sa_g1    = (const float*)d_in[20];
    const float* sa_b1    = (const float*)d_in[21];
    const float* sa_W1    = (const float*)d_in[22];
    const float* sa_c1    = (const float*)d_in[23];
    const float* sa_W2    = (const float*)d_in[24];
    const float* sa_c2    = (const float*)d_in[25];
    const float* sa_g2    = (const float*)d_in[26];
    const float* sa_b2    = (const float*)d_in[27];
    const float* ma_W1    = (const float*)d_in[28];
    const float* ma_b1    = (const float*)d_in[29];
    const float* ma_W2    = (const float*)d_in[30];
    const float* ma_b2    = (const float*)d_in[31];
    const float* f_W1     = (const float*)d_in[32];
    const float* f_b1     = (const float*)d_in[33];
    const float* f_W2     = (const float*)d_in[34];
    const float* f_b2     = (const float*)d_in[35];
    const int*   node_idx = (const int*)d_in[36];
    const int*   edge_idx = (const int*)d_in[37];
    const float* time_idx = (const float*)d_in[38];
    const int*   src_idx  = (const int*)d_in[39];
    const int*   tgt_idx  = (const int*)d_in[40];

    char* base = (char*)d_ws;
    size_t off = 0;
    auto alloc = [&](size_t bytes) -> void* {
        void* p = base + off;
        off = (off + bytes + 255) & ~(size_t)255;
        return p;
    };

    // --- packed weights / biases ---
    bf16*  wqkv_p   = (bf16*)alloc((size_t)QKVP * DP * 2);
    float* bqkv_p   = (float*)alloc((size_t)QKVP * 4);
    bf16*  wo_p     = (bf16*)alloc((size_t)DP * DP * 2);
    float* bo_p     = (float*)alloc((size_t)DP * 4);
    bf16*  w1_p     = (bf16*)alloc((size_t)FFND * DP * 2);
    bf16*  w2_p     = (bf16*)alloc((size_t)DP * FFND * 2);
    float* c2_p     = (float*)alloc((size_t)DP * 4);
    bf16*  maw1_p   = (bf16*)alloc((size_t)128 * DP * 2);
    bf16*  maw2_p   = (bf16*)alloc((size_t)128 * 128 * 2);
    bf16*  sawqkv_p = (bf16*)alloc((size_t)384 * 128 * 2);
    bf16*  sawo_p   = (bf16*)alloc((size_t)128 * 128 * 2);
    bf16*  saw1_p   = (bf16*)alloc((size_t)FFND * 128 * 2);
    bf16*  saw2_p   = (bf16*)alloc((size_t)128 * FFND * 2);
    bf16*  fw1_p    = (bf16*)alloc((size_t)128 * FPP * 2);

    // --- activations ---
    float* X32   = (float*)alloc((size_t)NT * D0 * 4);       // later aliased as h2
    bf16*  Xb    = (bf16*)alloc((size_t)NT * DP * 2);
    float* qkv   = (float*)alloc((size_t)NT * QKVP * 4);     // later aliased as f1b / sf1b
    bf16*  ao    = (bf16*)alloc((size_t)NT * DP * 2);
    float* proj  = (float*)alloc((size_t)NT * DP * 4);       // later aliased as ffn2
    float* h1_32 = (float*)alloc((size_t)NT * D0 * 4);
    bf16*  h1_b  = (bf16*)alloc((size_t)NT * DP * 2);
    bf16*  encb  = (bf16*)alloc((size_t)NS * DP * 2);
    bf16*  ma1b  = (bf16*)alloc((size_t)NS * 128 * 2);
    float* g32   = (float*)alloc((size_t)NS * 128 * 4);
    bf16*  gb    = (bf16*)alloc((size_t)NS * 128 * 2);
    float* sqkv  = (float*)alloc((size_t)NS * 384 * 4);
    bf16*  saob  = (bf16*)alloc((size_t)NS * 128 * 2);
    float* sproj = (float*)alloc((size_t)NS * 128 * 4);
    float* t32   = (float*)alloc((size_t)NS * 128 * 4);
    bf16*  tb    = (bf16*)alloc((size_t)NS * 128 * 2);
    float* sffn2 = (float*)alloc((size_t)NS * 128 * 4);
    float* gfin  = (float*)alloc((size_t)NS * 128 * 4);
    bf16*  featsb= (bf16*)alloc((size_t)NS * FPP * 2);
    float* ff1   = (float*)alloc((size_t)NS * 128 * 4);

    // aliases (lifetimes disjoint)
    bf16*  f1b  = (bf16*)qkv;    // [NT, 4096] bf16 (157MB <= 162MB)
    float* ffn2 = proj;          // [NT, 704]  f32
    float* h2   = X32;           // [NT, 688]  f32
    bf16*  sf1b = (bf16*)qkv;    // [NS, 4096] bf16

    auto gemm = [&](const bf16* A, const bf16* Wm, const float* bias,
                    float* oF, bf16* oB, int M, int N, int K, int relu) {
        dim3 g(M / 128, N / 64);
        gemm_bf16<<<g, 256, 0, stream>>>(A, Wm, bias, oF, oB, M, N, K, relu);
    };

    // ---- weight prepack ----
    pack_w<<<QKVP, 256, 0, stream>>>(cat_Wqkv, wqkv_p, QKV0, D0, DP);
    pack_bias<<<(QKVP + 255) / 256, 256, 0, stream>>>(cat_bqkv, bqkv_p, QKV0, QKVP);
    pack_w<<<DP, 256, 0, stream>>>(cat_Wo, wo_p, D0, D0, DP);
    pack_bias<<<(DP + 255) / 256, 256, 0, stream>>>(cat_bo, bo_p, D0, DP);
    pack_w<<<FFND, 256, 0, stream>>>(cat_W1, w1_p, FFND, D0, DP);
    pack_w<<<DP, 256, 0, stream>>>(cat_W2, w2_p, D0, FFND, FFND);
    pack_bias<<<(DP + 255) / 256, 256, 0, stream>>>(cat_c2, c2_p, D0, DP);
    pack_w<<<128, 256, 0, stream>>>(ma_W1, maw1_p, 128, D0, DP);
    pack_w<<<128, 256, 0, stream>>>(ma_W2, maw2_p, 128, 128, 128);
    pack_w<<<384, 256, 0, stream>>>(sa_Wqkv, sawqkv_p, 384, 128, 128);
    pack_w<<<128, 256, 0, stream>>>(sa_Wo, sawo_p, 128, 128, 128);
    pack_w<<<FFND, 256, 0, stream>>>(sa_W1, saw1_p, FFND, 128, 128);
    pack_w<<<128, 256, 0, stream>>>(sa_W2, saw2_p, 128, FFND, FFND);
    pack_w<<<128, 256, 0, stream>>>(f_W1, fw1_p, 128, FP0, FPP);

    // ---- stage 1: build X ----
    build_x<<<NT, 256, 0, stream>>>(node_table, edge_table, basis_freq, phase,
                                    node_idx, edge_idx, time_idx, X32, Xb);

    // ---- cat encoder ----
    gemm(Xb, wqkv_p, bqkv_p, qkv, nullptr, NT, QKVP, DP, 0);
    cat_attn<<<NS * 8, 32, 0, stream>>>(qkv, ao);
    gemm(ao, wo_p, bo_p, proj, nullptr, NT, DP, DP, 0);
    add_ln<<<NT, 256, 0, stream>>>(X32, D0, proj, DP, cat_g1, cat_b1, D0,
                                   h1_32, h1_b, DP);
    gemm(h1_b, w1_p, cat_c1, nullptr, f1b, NT, FFND, DP, 1);
    gemm(f1b, w2_p, c2_p, ffn2, nullptr, NT, DP, FFND, 0);
    add_ln<<<NT, 256, 0, stream>>>(h1_32, D0, ffn2, DP, cat_g2, cat_b2, D0,
                                   h2, nullptr, 0);
    mean3<<<NS, 256, 0, stream>>>(h2, encb);

    // ---- ma MLP ----
    gemm(encb, maw1_p, ma_b1, nullptr, ma1b, NS, 128, DP, 1);
    gemm(ma1b, maw2_p, ma_b2, g32, gb, NS, 128, 128, 0);

    // ---- sa encoder ----
    gemm(gb, sawqkv_p, sa_bqkv, sqkv, nullptr, NS, 384, 128, 0);
    sa_attn<<<512, 128, 0, stream>>>(sqkv, saob);
    gemm(saob, sawo_p, sa_bo, sproj, nullptr, NS, 128, 128, 0);
    add_ln<<<NS, 256, 0, stream>>>(g32, 128, sproj, 128, sa_g1, sa_b1, 128,
                                   t32, tb, 128);
    gemm(tb, saw1_p, sa_c1, nullptr, sf1b, NS, FFND, 128, 1);
    gemm(sf1b, saw2_p, sa_c2, sffn2, nullptr, NS, 128, FFND, 0);
    add_ln<<<NS, 256, 0, stream>>>(t32, 128, sffn2, 128, sa_g2, sa_b2, 128,
                                   gfin, nullptr, 0);

    // ---- final head ----
    build_feats<<<NS, 128, 0, stream>>>(gfin, node_table, src_idx, tgt_idx, featsb);
    gemm(featsb, fw1_p, f_b1, ff1, nullptr, NS, 128, FPP, 1);
    final_head<<<NS, 128, 0, stream>>>(ff1, f_W2, f_b2, (float*)d_out);
}
